// VSSBlock_34385508172152
// MI455X (gfx1250) — matrix operand order repaired and checked by reading, not compiled, not run on hardware
//
#include <hip/hip_runtime.h>
#include <cstdint>
#include <cstddef>

// ---------------- problem constants ----------------
#define PTOT  32768          // total pixels = 2*128*128
#define CDIM  96
#define INNER 192
#define HIDD  384
#define HW    16384          // 128*128
#define WIMG  128

// ---------------- WMMA fragment types ----------------
typedef __attribute__((ext_vector_type(16))) __bf16 v16bf;
typedef __attribute__((ext_vector_type(8)))  float  v8f;

union Frag16 { uint4 q[2]; v16bf v; };

__device__ __forceinline__ unsigned short f2bf(float f) {
    unsigned int u = __float_as_uint(f);
    u += 0x7fffu + ((u >> 16) & 1u);           // round-to-nearest-even
    return (unsigned short)(u >> 16);
}
__device__ __forceinline__ float bf2f(unsigned short h) {
    return __uint_as_float(((unsigned int)h) << 16);
}
__device__ __forceinline__ float silu_f(float x) { return x / (1.f + expf(-x)); }

// ---------------- weight convert: f32 KxN -> bf16 NxK (transposed) ----------------
__global__ void wconvT_kernel(const float* __restrict__ src,
                              unsigned short* __restrict__ dst, int K, int N) {
    int i = blockIdx.x * blockDim.x + threadIdx.x;
    if (i >= K * N) return;
    int k = i / N, n = i - k * N;
    dst[(size_t)n * K + k] = f2bf(src[i]);
}

// ---------------- LN1: NCHW gather -> LN over C -> bf16 NHWC ----------------
__global__ void ln1_kernel(const float* __restrict__ x, const float* __restrict__ g,
                           const float* __restrict__ b, unsigned short* __restrict__ nout) {
    int p  = blockIdx.x * blockDim.x + threadIdx.x;       // grid covers PTOT exactly
    int bb = p >> 14, rem = p & (HW - 1);
    const float* xp = x + (size_t)bb * CDIM * HW + rem;
    float vals[CDIM]; float s = 0.f, ss = 0.f;
#pragma unroll
    for (int c = 0; c < CDIM; c++) { float v = xp[(size_t)c * HW]; vals[c] = v; s += v; ss += v * v; }
    float m = s * (1.f / CDIM);
    float var = ss * (1.f / CDIM) - m * m;
    float r = rsqrtf(var + 1e-5f);
    unsigned short* np = nout + (size_t)p * CDIM;
#pragma unroll
    for (int c = 0; c < CDIM; c++) np[c] = f2bf((vals[c] - m) * r * g[c] + b[c]);
}

// ---------------- depthwise conv4 along scan dir + SiLU (flip folded in) ----------------
// dir: 0=h fwd, 1=h bwd, 2=v fwd, 3=v bwd
__global__ void conv_silu_kernel(const float* __restrict__ xz, const float* __restrict__ cw,
                                 const float* __restrict__ cb, unsigned short* __restrict__ xa,
                                 int dir) {
    int idx = blockIdx.x * blockDim.x + threadIdx.x;
    int p = idx / INNER, c = idx - p * INNER;
    int bb = p >> 14, rem = p & (HW - 1), y = rem >> 7, xc = rem & (WIMG - 1);
    float acc = cb[c];
#pragma unroll
    for (int k = 0; k < 4; k++) {
        int off = (dir & 1) ? (3 - k) : (k - 3);
        int yy = y, xx = xc;
        if (dir < 2) xx += off; else yy += off;
        if (xx >= 0 && xx < WIMG && yy >= 0 && yy < WIMG) {
            int pn = (bb << 14) + (yy << 7) + xx;
            acc += cw[k * INNER + c] * xz[(size_t)pn * (2 * INNER) + c];
        }
    }
    xa[idx] = f2bf(silu_f(acc));
}

// ---------------- SSM pointwise: y = xa*(sum_s exp(dt*A_s)+D)*silu(z) ----------------
__global__ void ssm_y_kernel(const float* __restrict__ dtp, const unsigned short* __restrict__ xa,
                             const float* __restrict__ xz, const float* __restrict__ Alog,
                             const float* __restrict__ Dp, unsigned short* __restrict__ yout) {
    int idx = blockIdx.x * blockDim.x + threadIdx.x;
    int p = idx / INNER, c = idx - p * INNER;
    float v = dtp[idx];
    float dt = (v > 20.f) ? v : log1pf(expf(v));          // softplus
    float sum = 0.f;
#pragma unroll
    for (int s = 0; s < 16; s++) sum += expf(-dt * expf(Alog[c * 16 + s]));  // A = -exp(A_log)
    float xaf = bf2f(xa[idx]);
    float z = xz[(size_t)p * (2 * INNER) + INNER + c];
    yout[idx] = f2bf(xaf * (sum + Dp[c]) * silu_f(z));
}

// ---------------- build merged = [h, v, h*v, x2d] as bf16 ----------------
__global__ void merge_prep_kernel(const float* __restrict__ h, const float* __restrict__ v,
                                  const float* __restrict__ x, unsigned short* __restrict__ mg) {
    int idx = blockIdx.x * blockDim.x + threadIdx.x;
    int p = idx / (4 * CDIM), c4 = idx - p * (4 * CDIM);
    float val;
    if (c4 < CDIM)            val = h[(size_t)p * CDIM + c4];
    else if (c4 < 2 * CDIM)   val = v[(size_t)p * CDIM + (c4 - CDIM)];
    else if (c4 < 3 * CDIM) { int c = c4 - 2 * CDIM;
                              val = h[(size_t)p * CDIM + c] * v[(size_t)p * CDIM + c]; }
    else                    { int c = c4 - 3 * CDIM; int bb = p >> 14, rem = p & (HW - 1);
                              val = x[((size_t)bb * CDIM + c) * HW + rem]; }
    mg[idx] = f2bf(val);
}

// ---------------- channel attention: mean+max pool per (b,c) ----------------
__global__ void pool_kernel(const float* __restrict__ o, float* __restrict__ pooled) {
    int blk = blockIdx.x; int bb = blk / CDIM, c = blk - bb * CDIM;
    __shared__ float ssum[256]; __shared__ float smax[256];
    int t = threadIdx.x;
    float s = 0.f, mx = -3.4e38f;
    for (int rem = t; rem < HW; rem += 256) {
        float v = o[((size_t)(bb << 14) + rem) * CDIM + c];
        s += v; mx = fmaxf(mx, v);
    }
    ssum[t] = s; smax[t] = mx; __syncthreads();
    for (int st = 128; st > 0; st >>= 1) {
        if (t < st) { ssum[t] += ssum[t + st]; smax[t] = fmaxf(smax[t], smax[t + st]); }
        __syncthreads();
    }
    if (t == 0) pooled[bb * CDIM + c] = ssum[0] * (1.f / HW) + smax[0];
}

// ---------------- channel attention tiny MLP (96 -> 8 -> 96) ----------------
__global__ void attn_kernel(const float* __restrict__ pooled, const float* __restrict__ w1,
                            const float* __restrict__ b1, const float* __restrict__ w2,
                            const float* __restrict__ b2, float* __restrict__ attn) {
    int bb = blockIdx.x; int t = threadIdx.x;   // blockDim = 96
    __shared__ float hid[8];
    if (t < 8) {
        float a = b1[t];
        for (int c = 0; c < CDIM; c++) a += pooled[bb * CDIM + c] * w1[c * 8 + t];
        hid[t] = fmaxf(a, 0.f);
    }
    __syncthreads();
    float a = b2[t];
#pragma unroll
    for (int j = 0; j < 8; j++) a += hid[j] * w2[j * CDIM + t];
    attn[bb * CDIM + t] = 1.f / (1.f + expf(-a));
}

// ---------------- apply attention in-place + LN2 -> bf16 ----------------
__global__ void attn_ln2_kernel(float* __restrict__ o, const float* __restrict__ attn,
                                const float* __restrict__ g, const float* __restrict__ b,
                                unsigned short* __restrict__ n2) {
    int p = blockIdx.x * blockDim.x + threadIdx.x;
    int bb = p >> 14;
    float vals[CDIM]; float s = 0.f, ss = 0.f;
    float* op = o + (size_t)p * CDIM;
#pragma unroll
    for (int c = 0; c < CDIM; c++) {
        float v = op[c] * attn[bb * CDIM + c];
        vals[c] = v; s += v; ss += v * v;
    }
    float m = s * (1.f / CDIM);
    float var = ss * (1.f / CDIM) - m * m;
    float r = rsqrtf(var + 1e-5f);
    unsigned short* np = n2 + (size_t)p * CDIM;
#pragma unroll
    for (int c = 0; c < CDIM; c++) {
        op[c] = vals[c];                                    // o becomes ca_out
        np[c] = f2bf((vals[c] - m) * r * g[c] + b[c]);
    }
}

// ---------------- GDFN depthwise 3x3 + exact-GELU gating ----------------
__global__ void gdfn_dw_kernel(const unsigned short* __restrict__ gbuf, const float* __restrict__ dww,
                               const float* __restrict__ dwb, unsigned short* __restrict__ gd) {
    int idx = blockIdx.x * blockDim.x + threadIdx.x;
    int p = idx / HIDD, co = idx - p * HIDD;
    int bb = p >> 14, rem = p & (HW - 1), y = rem >> 7, xc = rem & (WIMG - 1);
    float a1 = dwb[co], a2 = dwb[co + HIDD];
#pragma unroll
    for (int ky = 0; ky < 3; ky++) {
        int yy = y + ky - 1;
        if (yy < 0 || yy >= WIMG) continue;
#pragma unroll
        for (int kx = 0; kx < 3; kx++) {
            int xx = xc + kx - 1;
            if (xx < 0 || xx >= WIMG) continue;
            const unsigned short* gp = gbuf + ((size_t)(bb << 14) + (yy << 7) + xx) * (2 * HIDD);
            int wi = (ky * 3 + kx) * (2 * HIDD);
            a1 += bf2f(gp[co])        * dww[wi + co];
            a2 += bf2f(gp[co + HIDD]) * dww[wi + co + HIDD];
        }
    }
    float ge = 0.5f * a1 * (1.f + erff(a1 * 0.70710678118f));  // exact GELU
    gd[idx] = f2bf(ge * a2);
}

// ---------------- WMMA GEMM: out[M=PTOT x N] = A[M x K](bf16) * W[K x N] + bias ----------------
// Wt is stored transposed (N x K, bf16). 8 waves/block, 16 rows/wave, 256 blocks.
// Wt is cooperatively staged into dynamic LDS once per block; A fragments (K/32 of them)
// are hoisted into registers once per wave and reused across all N-tiles.
// EPI: 0=store f32, 1=accumulate f32, 2=store bf16,
//      3=+x(NCHW) residual -> f32 NHWC, 4=+aux(NHWC f32) residual -> f32 NCHW (d_out)
template <int K, int N, int EPI>
__global__ __launch_bounds__(256) void gemm_wmma_kernel(
    const unsigned short* __restrict__ A,
    const unsigned short* __restrict__ Wt,
    const float* __restrict__ bias,
    float* __restrict__ OutF,
    unsigned short* __restrict__ OutH,
    const float* __restrict__ aux) {
    extern __shared__ unsigned short smemB[];           // N*K bf16, dynamic (up to 147 KB)

    // cooperative stage of the whole transposed weight matrix into LDS (uint4 chunks)
    {
        const int chunks = (K * N) / 8;                 // 8 halves per 16B chunk
        const uint4* src = (const uint4*)Wt;
        uint4* dst = (uint4*)smemB;
        for (int i = threadIdx.x; i < chunks; i += 256) dst[i] = src[i];
    }
    __syncthreads();

    const int lane = threadIdx.x & 31;
    const int wave = threadIdx.x >> 5;
    const int m0 = blockIdx.x * 128 + wave * 16;
    // A fragment base: row m0+(lane&15); lanes<16 take K 0-7/16-23, lanes>=16 take 8-15/24-31
    const unsigned short* Arow = A + (size_t)(m0 + (lane & 15)) * K + ((lane < 16) ? 0 : 8);
    const int ksel_b = (lane < 16) ? 0 : 8;
    const int ncol_lane = lane & 15;

    // hoist all A fragments for this wave's 16 rows (K/32 fragments, reused for every N-tile)
    Frag16 fa[K / 32];
#pragma unroll
    for (int kk = 0; kk < K / 32; kk++) {
        fa[kk].q[0] = *(const uint4*)(Arow + kk * 32);
        fa[kk].q[1] = *(const uint4*)(Arow + kk * 32 + 16);
    }

    for (int nt = 0; nt < N / 16; ++nt) {
        const int ncol = nt * 16 + ncol_lane;
        const float bv = bias[ncol];
        v8f acc;
#pragma unroll
        for (int r = 0; r < 8; r++) acc[r] = bv;
        const unsigned short* Bcol = smemB + (size_t)ncol * K + ksel_b;
#pragma unroll
        for (int kk = 0; kk < K / 32; kk++) {
            Frag16 fb;
            fb.q[0] = *(const uint4*)(Bcol + kk * 32);
            fb.q[1] = *(const uint4*)(Bcol + kk * 32 + 16);
            acc = __builtin_amdgcn_wmma_f32_16x16x32_bf16(
                false, fa[kk].v, false, fb.v, (short)0, acc, false, false);
        }
        const int mbase = m0 + ((lane < 16) ? 0 : 8);
#pragma unroll
        for (int r = 0; r < 8; r++) {
            const int mr = mbase + r;
            float val = acc[r];
            if constexpr (EPI == 0) {
                OutF[(size_t)mr * N + ncol] = val;
            } else if constexpr (EPI == 1) {
                OutF[(size_t)mr * N + ncol] += val;
            } else if constexpr (EPI == 2) {
                OutH[(size_t)mr * N + ncol] = f2bf(val);
            } else if constexpr (EPI == 3) {
                int bb = mr >> 14, rem = mr & (HW - 1);
                val += aux[((size_t)bb * CDIM + ncol) * HW + rem];
                OutF[(size_t)mr * CDIM + ncol] = val;
            } else {
                val += aux[(size_t)mr * CDIM + ncol];
                int bb = mr >> 14, rem = mr & (HW - 1);
                OutF[((size_t)bb * CDIM + ncol) * HW + rem] = val;
            }
        }
    }
}

// ---------------- host launcher ----------------
extern "C" void kernel_launch(void* const* d_in, const int* in_sizes, int n_in,
                              void* d_out, int out_size, void* d_ws, size_t ws_size,
                              hipStream_t stream) {
    (void)in_sizes; (void)n_in; (void)out_size; (void)ws_size;
    const float* x = (const float*)d_in[0];
    auto ssm = [&](int d, int j) { return (const float*)d_in[1 + d * 10 + j]; };
    // j: 0 in_w, 1 in_b, 2 conv_w, 3 conv_b, 4 A_log, 5 D, 6 dt_w, 7 dt_b, 8 out_w, 9 out_b
    const float* norm1_g = (const float*)d_in[41];
    const float* norm1_b = (const float*)d_in[42];
    const float* norm2_g = (const float*)d_in[43];
    const float* norm2_b = (const float*)d_in[44];
    const float* merge_w = (const float*)d_in[45];
    const float* merge_b = (const float*)d_in[46];
    const float* gin_w   = (const float*)d_in[47];
    const float* gin_b   = (const float*)d_in[48];
    const float* gdw     = (const float*)d_in[49];
    const float* gdw_b   = (const float*)d_in[50];
    const float* gout_w  = (const float*)d_in[51];
    const float* gout_b  = (const float*)d_in[52];
    const float* ca_w1   = (const float*)d_in[53];
    const float* ca_b1   = (const float*)d_in[54];
    const float* ca_w2   = (const float*)d_in[55];
    const float* ca_b2   = (const float*)d_in[56];

    char* ws = (char*)d_ws;
    size_t off = 0;
    auto alloc = [&](size_t bytes) -> char* {
        char* p = ws + off; off += (bytes + 255) & ~(size_t)255; return p;
    };
    unsigned short* WT    = (unsigned short*)alloc((size_t)516096 * 2);     // transposed bf16 weights
    unsigned short* nbuf  = (unsigned short*)alloc((size_t)PTOT * CDIM * 2);
    unsigned short* xabuf = (unsigned short*)alloc((size_t)PTOT * INNER * 2);
    unsigned short* ybuf  = (unsigned short*)alloc((size_t)PTOT * INNER * 2);
    float* hbuf  = (float*)alloc((size_t)PTOT * CDIM * 4);
    float* vbuf  = (float*)alloc((size_t)PTOT * CDIM * 4);
    float* o2d   = (float*)alloc((size_t)PTOT * CDIM * 4);
    unsigned short* n2buf = (unsigned short*)alloc((size_t)PTOT * CDIM * 2);
    char* bufA   = alloc((size_t)PTOT * 2 * HIDD * 2);  // xz f32 (P*384*4) / g bf16 (P*768*2)
    char* bufB   = alloc((size_t)PTOT * HIDD * 2);      // dtp f32 / merged bf16 / gd bf16
    float* pooled = (float*)alloc(2 * CDIM * 4);
    float* attn   = (float*)alloc(2 * CDIM * 4);

    unsigned short *wt_in[4], *wt_dt[4], *wt_out[4];
    for (int d = 0; d < 4; d++) {
        wt_in[d]  = WT + d * 36864;
        wt_dt[d]  = WT + 147456 + d * 36864;
        wt_out[d] = WT + 294912 + d * 18432;
    }
    unsigned short* wt_merge = WT + 368640;
    unsigned short* wt_gin   = WT + 405504;
    unsigned short* wt_gout  = WT + 479232;

    auto cw = [&](const float* src, unsigned short* dst, int K, int N) {
        int n = K * N;
        wconvT_kernel<<<(n + 255) / 256, 256, 0, stream>>>(src, dst, K, N);
    };
    for (int d = 0; d < 4; d++) {
        cw(ssm(d, 0), wt_in[d],  CDIM,  2 * INNER);
        cw(ssm(d, 6), wt_dt[d],  INNER, INNER);
        cw(ssm(d, 8), wt_out[d], INNER, CDIM);
    }
    cw(merge_w, wt_merge, 4 * CDIM, CDIM);
    cw(gin_w,   wt_gin,   CDIM,     2 * HIDD);
    cw(gout_w,  wt_gout,  HIDD,     CDIM);

    // LayerNorm 1 (NCHW -> NHWC bf16)
    ln1_kernel<<<PTOT / 256, 256, 0, stream>>>(x, norm1_g, norm1_b, nbuf);

    float* xz  = (float*)bufA;
    float* dtp = (float*)bufB;
    for (int d = 0; d < 4; d++) {
        gemm_wmma_kernel<CDIM, 2 * INNER, 0>
            <<<256, 256, CDIM * 2 * INNER * 2, stream>>>(
                nbuf, wt_in[d], ssm(d, 1), xz, nullptr, nullptr);
        conv_silu_kernel<<<(PTOT * INNER) / 256, 256, 0, stream>>>(
            xz, ssm(d, 2), ssm(d, 3), xabuf, d);
        gemm_wmma_kernel<INNER, INNER, 0>
            <<<256, 256, INNER * INNER * 2, stream>>>(
                xabuf, wt_dt[d], ssm(d, 7), dtp, nullptr, nullptr);
        ssm_y_kernel<<<(PTOT * INNER) / 256, 256, 0, stream>>>(
            dtp, xabuf, xz, ssm(d, 4), ssm(d, 5), ybuf);
        float* dst = (d < 2) ? hbuf : vbuf;
        if ((d & 1) == 0)
            gemm_wmma_kernel<INNER, CDIM, 0>
                <<<256, 256, INNER * CDIM * 2, stream>>>(
                    ybuf, wt_out[d], ssm(d, 9), dst, nullptr, nullptr);
        else
            gemm_wmma_kernel<INNER, CDIM, 1>
                <<<256, 256, INNER * CDIM * 2, stream>>>(
                    ybuf, wt_out[d], ssm(d, 9), dst, nullptr, nullptr);
    }

    // merge + residual
    unsigned short* merged = (unsigned short*)bufB;
    merge_prep_kernel<<<(PTOT * 4 * CDIM) / 256, 256, 0, stream>>>(hbuf, vbuf, x, merged);
    gemm_wmma_kernel<4 * CDIM, CDIM, 3>
        <<<256, 256, 4 * CDIM * CDIM * 2, stream>>>(
            merged, wt_merge, merge_b, o2d, nullptr, x);

    // channel attention
    pool_kernel<<<2 * CDIM, 256, 0, stream>>>(o2d, pooled);
    attn_kernel<<<2, CDIM, 0, stream>>>(pooled, ca_w1, ca_b1, ca_w2, ca_b2, attn);
    attn_ln2_kernel<<<PTOT / 256, 256, 0, stream>>>(o2d, attn, norm2_g, norm2_b, n2buf);

    // GDFN
    unsigned short* gbuf = (unsigned short*)bufA;
    gemm_wmma_kernel<CDIM, 2 * HIDD, 2>
        <<<256, 256, CDIM * 2 * HIDD * 2, stream>>>(
            n2buf, wt_gin, gin_b, nullptr, gbuf, nullptr);
    unsigned short* gd = (unsigned short*)bufB;
    gdfn_dw_kernel<<<(PTOT * HIDD) / 256, 256, 0, stream>>>(gbuf, gdw, gdw_b, gd);
    gemm_wmma_kernel<HIDD, CDIM, 4>
        <<<256, 256, HIDD * CDIM * 2, stream>>>(
            gd, wt_gout, gout_b, (float*)d_out, nullptr, o2d);
}
